// RLGHGTLayer_17300128269022
// MI455X (gfx1250) — matrix operand order repaired and checked by reading, not compiled, not run on hardware
//
#include <hip/hip_runtime.h>
#include <hip/hip_bf16.h>
#include <math.h>

#define F 128
#define HEADS 8
#define HD 16
#define EPS 1e-5f

typedef __bf16 v16bf __attribute__((ext_vector_type(16)));
typedef float  v8f   __attribute__((ext_vector_type(8)));

union Frag { v16bf v; unsigned int u[8]; };

__device__ __forceinline__ unsigned short f2bf(float f) {
    unsigned u = __float_as_uint(f);
    u += 0x7FFFu + ((u >> 16) & 1u);           // round-to-nearest-even
    return (unsigned short)(u >> 16);
}
__device__ __forceinline__ unsigned enc_ord(float f) {  // order-preserving uint encode
    unsigned u = __float_as_uint(f);
    return (u & 0x80000000u) ? ~u : (u | 0x80000000u);
}
__device__ __forceinline__ float dec_ord(unsigned u) {
    return (u & 0x80000000u) ? __uint_as_float(u ^ 0x80000000u) : __uint_as_float(~u);
}

template<int ACT> __device__ __forceinline__ float actf(float x) {
    if (ACT == 1) return x > 0.f ? x : 0.f;                           // relu
    if (ACT == 2) return 0.5f * x * (1.f + erff(x * 0.70710678118f)); // exact gelu
    if (ACT == 3) return 1.f / (1.f + __expf(-x));                    // sigmoid
    return x;
}

// ---------------------------------------------------------------------------
// Fused (optional LayerNorm) + GEMM via V_WMMA_F32_16X16X32_BF16.
// Block = 128 threads (4 waves). Block computes 16 rows x 128 cols of Y;
// blockIdx.y selects the 128-col slab for DO>128. X given as up to 4 segments
// of 128 cols each (implements concat inputs); ldx==0 broadcasts one row.
// ---------------------------------------------------------------------------
template<int DI, int ACT, bool LN>
__global__ __launch_bounds__(128)
void lngemm_kernel(const float* __restrict__ X0, int ldx0,
                   const float* __restrict__ X1, int ldx1,
                   const float* __restrict__ X2, int ldx2,
                   const float* __restrict__ X3, int ldx3,
                   const float* __restrict__ lng, const float* __restrict__ lnb,
                   const float* __restrict__ W, int ldw,
                   const float* __restrict__ c, const float* __restrict__ c2,
                   float* __restrict__ Y, int ldy, int nrows)
{
    __shared__ unsigned short As[16][DI + 8];   // bf16 activations (row-major)
    __shared__ unsigned short Bs[128][32];      // bf16 weights, [col][k]
    __shared__ float redA[16][8], redB[16][8];

    const int tid  = threadIdx.x;
    const int wave = tid >> 5;
    const int lane = tid & 31;
    const int r0   = blockIdx.x * 16;
    const int colbase = blockIdx.y * 128;

    if (LN) {  // DI==128 whenever LN is used
        const int row = tid >> 3, sub = tid & 7;
        const int gr = r0 + row;
        float x[16]; float s = 0.f, s2 = 0.f;
        #pragma unroll
        for (int j = 0; j < 16; ++j) {
            float v = (gr < nrows) ? X0[(size_t)gr * ldx0 + sub * 16 + j] : 0.f;
            x[j] = v; s += v; s2 += v * v;
        }
        redA[row][sub] = s; redB[row][sub] = s2;
        __syncthreads();
        float mu = 0.f, msq = 0.f;
        #pragma unroll
        for (int k = 0; k < 8; ++k) { mu += redA[row][k]; msq += redB[row][k]; }
        mu *= (1.f / DI); msq *= (1.f / DI);
        float rstd = rsqrtf(msq - mu * mu + EPS);
        #pragma unroll
        for (int j = 0; j < 16; ++j) {
            int col = sub * 16 + j;
            As[row][col] = f2bf((x[j] - mu) * rstd * lng[col] + lnb[col]);
        }
    } else {
        const float* Xs[4] = {X0, X1, X2, X3};
        const int    ls[4] = {ldx0, ldx1, ldx2, ldx3};
        for (int idx = tid; idx < 16 * DI; idx += 128) {
            int row = idx / DI, col = idx % DI;
            int seg = col >> 7, cc = col & 127;
            int gr = r0 + row;
            float v = (gr < nrows) ? Xs[seg][(size_t)gr * ls[seg] + cc] : 0.f;
            As[row][col] = f2bf(v);
        }
    }
    __syncthreads();

    v8f acc0 = {}; v8f acc1 = {};
    const int n0   = wave * 32;                 // this wave's 32 output cols
    const int arow = lane & 15;
    const int aklo = (lane < 16) ? 0 : 16;      // contiguous k = 16h + s (matches B)
    const int akhi = (lane < 16) ? 8 : 24;
    const int bcol = lane & 15;
    const int bk   = (lane < 16) ? 0 : 16;      // contiguous k = 16h + s (matches A)

    for (int k0 = 0; k0 < DI; k0 += 32) {
        __syncthreads();                        // prev iter fragments consumed
        #pragma unroll
        for (int kk = 0; kk < 32; ++kk)
            Bs[tid][kk] = f2bf(W[(size_t)(k0 + kk) * ldw + colbase + tid]);
        __syncthreads();

        Frag a, b0, b1;
        #pragma unroll
        for (int i = 0; i < 4; ++i) a.u[i]     = *(const unsigned*)&As[arow][k0 + aklo + 2 * i];
        #pragma unroll
        for (int i = 0; i < 4; ++i) a.u[4 + i] = *(const unsigned*)&As[arow][k0 + akhi + 2 * i];
        #pragma unroll
        for (int i = 0; i < 8; ++i) b0.u[i] = *(const unsigned*)&Bs[n0 + bcol][bk + 2 * i];
        #pragma unroll
        for (int i = 0; i < 8; ++i) b1.u[i] = *(const unsigned*)&Bs[n0 + 16 + bcol][bk + 2 * i];

        acc0 = __builtin_amdgcn_wmma_f32_16x16x32_bf16(false, a.v, false, b0.v, (short)0, acc0, false, false);
        acc1 = __builtin_amdgcn_wmma_f32_16x16x32_bf16(false, a.v, false, b1.v, (short)0, acc1, false, false);
    }

    // C/D: vgpr i -> row i (lanes 0-15) or 8+i (lanes 16-31), col = lane&15
    const int crow = (lane < 16) ? 0 : 8;
    const int ccol = lane & 15;
    #pragma unroll
    for (int i = 0; i < 8; ++i) {
        int gr = r0 + crow + i;
        if (gr < nrows) {
            int col0 = colbase + n0 + ccol;
            int col1 = colbase + n0 + 16 + ccol;
            float v0 = acc0[i] + c[col0] + (c2 ? c2[col0] : 0.f);
            float v1 = acc1[i] + c[col1] + (c2 ? c2[col1] : 0.f);
            Y[(size_t)gr * ldy + col0] = actf<ACT>(v0);
            Y[(size_t)gr * ldy + col1] = actf<ACT>(v1);
        }
    }
}

// ----------------------------- edge kernels --------------------------------
__global__ void edge_score_kernel(const float* __restrict__ kmat, const float* __restrict__ qmat,
                                  const int* __restrict__ src, const int* __restrict__ dst,
                                  const float* __restrict__ bias, const float* __restrict__ scale,
                                  float* __restrict__ sc, unsigned* __restrict__ mbuf, int E)
{
    int idx = blockIdx.x * blockDim.x + threadIdx.x;
    if (idx >= E * HEADS) return;
    int e = idx >> 3, h = idx & 7;
    int s = src[e], d = dst[e];
    const float* kp = kmat + (size_t)s * F + h * HD;
    const float* qp = qmat + (size_t)d * F + h * HD;
    float acc = 0.f;
    #pragma unroll
    for (int j = 0; j < HD; ++j) acc += kp[j] * qp[j];
    float scl = 1.f + 1.f / (1.f + __expf(-scale[0]));
    float v = acc * 0.25f * scl + bias[h];
    sc[idx] = v;
    atomicMax(&mbuf[(size_t)d * HEADS + h], enc_ord(v));
}

__global__ void edge_exp_kernel(const float* __restrict__ sc, const int* __restrict__ dst,
                                const unsigned* __restrict__ mbuf,
                                float* __restrict__ ex, float* __restrict__ zbuf, int E)
{
    int idx = blockIdx.x * blockDim.x + threadIdx.x;
    if (idx >= E * HEADS) return;
    int e = idx >> 3, h = idx & 7;
    int d = dst[e];
    float v = __expf(sc[idx] - dec_ord(mbuf[(size_t)d * HEADS + h]));
    ex[idx] = v;
    atomicAdd(&zbuf[(size_t)d * HEADS + h], v);
}

__global__ void edge_agg_kernel(const float* __restrict__ vmat, const float* __restrict__ ex,
                                const float* __restrict__ zbuf,
                                const int* __restrict__ src, const int* __restrict__ dst,
                                float* __restrict__ out, int E)
{
    long long idx = (long long)blockIdx.x * blockDim.x + threadIdx.x;
    if (idx >= (long long)E * F) return;
    int e = (int)(idx >> 7), col = (int)(idx & 127), h = col >> 4;
    int s = src[e], d = dst[e];
    float a = ex[(size_t)e * HEADS + h] / zbuf[(size_t)d * HEADS + h];
    atomicAdd(&out[(size_t)d * F + col], vmat[(size_t)s * F + col] * a);
}

__global__ void meta_score_kernel(const float* __restrict__ sh, const float* __restrict__ tq,
                                  const int* __restrict__ src, const int* __restrict__ dst,
                                  float* __restrict__ ms, unsigned* __restrict__ mm, int E)
{
    int wave = threadIdx.x >> 5, lane = threadIdx.x & 31;
    int e = blockIdx.x * (blockDim.x >> 5) + wave;
    if (e >= E) return;
    int s = src[e], d = dst[e];
    float acc = 0.f;
    #pragma unroll
    for (int j = 0; j < 4; ++j) {
        int col = lane * 4 + j;
        acc += tanhf(sh[(size_t)s * F + col] + tq[(size_t)d * F + col]);
    }
    #pragma unroll
    for (int off = 16; off; off >>= 1) acc += __shfl_xor(acc, off, 32);
    if (lane == 0) {
        float v = acc * (1.f / F);
        ms[e] = v;
        atomicMax(&mm[d], enc_ord(v));
    }
}

__global__ void meta_exp_kernel(const float* __restrict__ ms, const int* __restrict__ dst,
                                const unsigned* __restrict__ mm,
                                float* __restrict__ me, float* __restrict__ mz, int E)
{
    int e = blockIdx.x * blockDim.x + threadIdx.x;
    if (e >= E) return;
    int d = dst[e];
    float v = __expf(ms[e] - dec_ord(mm[d]));
    me[e] = v;
    atomicAdd(&mz[d], v);
}

__global__ void meta_agg_kernel(const float* __restrict__ sh, const float* __restrict__ me,
                                const float* __restrict__ mz,
                                const int* __restrict__ src, const int* __restrict__ dst,
                                float* __restrict__ out, int E)
{
    long long idx = (long long)blockIdx.x * blockDim.x + threadIdx.x;
    if (idx >= (long long)E * F) return;
    int e = (int)(idx >> 7), col = (int)(idx & 127);
    int s = src[e], d = dst[e];
    float a = me[e] / mz[d];
    atomicAdd(&out[(size_t)d * F + col], sh[(size_t)s * F + col] * a);
}

// ----------------------------- update helpers ------------------------------
__global__ void colsum_kernel(const float* __restrict__ X, float* __restrict__ gsum, int N)
{
    int col = threadIdx.x;
    float s = 0.f;
    for (int r = blockIdx.x; r < N; r += gridDim.x) s += X[(size_t)r * F + col];
    atomicAdd(&gsum[col], s);
}

__global__ void glob_kernel(const float* __restrict__ gsum, float invN,
                            const float* __restrict__ g, const float* __restrict__ b,
                            const float* __restrict__ W, const float* __restrict__ c,
                            float* __restrict__ outv)
{
    __shared__ float xs[F]; __shared__ float xh[F];
    int t = threadIdx.x;
    float m = gsum[t] * invN;
    xs[t] = m; __syncthreads();
    float mu = 0.f; for (int i = 0; i < F; ++i) mu += xs[i]; mu *= (1.f / F);
    float var = 0.f; for (int i = 0; i < F; ++i) { float d = xs[i] - mu; var += d * d; } var *= (1.f / F);
    float v = (m - mu) * rsqrtf(var + EPS) * g[t] + b[t];
    xh[t] = v; __syncthreads();
    float acc = c[t];
    for (int i = 0; i < F; ++i) acc += xh[i] * W[i * F + t];
    outv[t] = acc;
}

__global__ void gate4_kernel(const float* __restrict__ rg, const float* __restrict__ W,
                             const float* __restrict__ c, float* __restrict__ gate, int N)
{
    int n = blockIdx.x * blockDim.x + threadIdx.x;
    if (n >= N) return;
    float a0 = c[0], a1 = c[1], a2 = c[2], a3 = c[3];
    const float* r = rg + (size_t)n * F;
    for (int i = 0; i < F; ++i) {
        float x = r[i];
        a0 += x * W[i * 4 + 0]; a1 += x * W[i * 4 + 1];
        a2 += x * W[i * 4 + 2]; a3 += x * W[i * 4 + 3];
    }
    gate[(size_t)n * 4 + 0] = a0; gate[(size_t)n * 4 + 1] = a1;
    gate[(size_t)n * 4 + 2] = a2; gate[(size_t)n * 4 + 3] = a3;
}

__global__ void mix_ln_kernel(const float* __restrict__ h, const float* __restrict__ local,
                              const float* __restrict__ mg, const float* __restrict__ topo,
                              const float* __restrict__ globv, const float* __restrict__ gate,
                              const float* __restrict__ n1g, const float* __restrict__ n1b,
                              float* __restrict__ fused, int N)
{
    int row = blockIdx.x, t = threadIdx.x;
    __shared__ float xs[F];
    float g0 = gate[(size_t)row * 4 + 0], g1 = gate[(size_t)row * 4 + 1];
    float g2 = gate[(size_t)row * 4 + 2], g3 = gate[(size_t)row * 4 + 3];
    size_t o = (size_t)row * F + t;
    float x = h[o] + g0 * local[o] + g1 * mg[o] + g2 * topo[o] + g3 * globv[t];
    xs[t] = x; __syncthreads();
    float mu = 0.f; for (int i = 0; i < F; ++i) mu += xs[i]; mu *= (1.f / F);
    float var = 0.f; for (int i = 0; i < F; ++i) { float d = xs[i] - mu; var += d * d; } var *= (1.f / F);
    fused[o] = (x - mu) * rsqrtf(var + EPS) * n1g[t] + n1b[t];
}

__global__ void add_ln_kernel(const float* __restrict__ A, const float* __restrict__ B,
                              const float* __restrict__ g, const float* __restrict__ b,
                              float* __restrict__ out, int N)
{
    int row = blockIdx.x, t = threadIdx.x;
    __shared__ float xs[F];
    size_t o = (size_t)row * F + t;
    float x = A[o] + B[o];
    xs[t] = x; __syncthreads();
    float mu = 0.f; for (int i = 0; i < F; ++i) mu += xs[i]; mu *= (1.f / F);
    float var = 0.f; for (int i = 0; i < F; ++i) { float d = xs[i] - mu; var += d * d; } var *= (1.f / F);
    out[o] = (x - mu) * rsqrtf(var + EPS) * g[t] + b[t];
}

// ----------------------- parameter pointer mapping -------------------------
// JAX pytree flatten = dicts sorted by key. Node block leaf order (48 leaves):
// bg1{W,c} bg2{W,c} f1{W,b,c,g} f2{W,c} glob{W,b,c,g} k{W,b,c,g} mg{W,c}
// mq{W,b,c,g} mv{W,b,c,g} n1{b,g} n2{b,g} out{W,b,c,g} q{W,b,c,g}
// topo{W,b,c,g} v{W,b,c,g}
struct NP {
    const float *bg1W, *bg1c, *bg2W, *bg2c;
    const float *f1W, *f1b, *f1c, *f1g, *f2W, *f2c;
    const float *globW, *globb, *globc, *globg;
    const float *kW, *kb, *kc, *kg;
    const float *mgW, *mgc;
    const float *mqW, *mqb, *mqc, *mqg;
    const float *mvW, *mvb, *mvc, *mvg;
    const float *n1b, *n1g, *n2b, *n2g;
    const float *outW, *outb, *outc, *outg;
    const float *qW, *qb, *qc, *qg;
    const float *topoW, *topob, *topoc, *topog;
    const float *vW, *vb, *vc, *vg;
};
static NP loadNP(void* const* din, int base) {
    NP p;
    const float** f = (const float**)&p;
    for (int j = 0; j < 48; ++j) f[j] = (const float*)din[base + j];
    return p;
}

extern "C" void kernel_launch(void* const* d_in, const int* in_sizes, int n_in,
                              void* d_out, int out_size, void* d_ws, size_t ws_size,
                              hipStream_t stream)
{
    const float* h_u = (const float*)d_in[0];
    const float* h_i = (const float*)d_in[1];
    const int* r0s = (const int*)d_in[2]; const int* r0d = (const int*)d_in[3];
    const int* r1s = (const int*)d_in[4]; const int* r1d = (const int*)d_in[5];
    const int* mus = (const int*)d_in[6]; const int* mud = (const int*)d_in[7];
    const int* mis = (const int*)d_in[8]; const int* mid = (const int*)d_in[9];

    NP Pi = loadNP(d_in, 10);                        // params['item'] block (10..57)
    const float* relb = (const float*)d_in[58];      // rel.bias [2,8]
    const float* relk = (const float*)d_in[59];      // rel.k    [2,128]
    const float* rels = (const float*)d_in[60];      // rel.scale[2,1]
    const float* relv = (const float*)d_in[61];      // rel.v    [2,128]
    NP Pu = loadNP(d_in, 62);                        // params['user'] block (62..109)

    const int NU = in_sizes[0] / F, NI = in_sizes[1] / F;
    const int E0 = in_sizes[2], E1 = in_sizes[4], EU = in_sizes[6], EI = in_sizes[8];
    const int NM = NU > NI ? NU : NI;
    int Emax = E0; if (E1 > Emax) Emax = E1; if (EU > Emax) Emax = EU; if (EI > Emax) Emax = EI;
    const size_t NFu = (size_t)NU * F, NFi = (size_t)NI * F;

    // ---- workspace layout ----
    float* base = (float*)d_ws; size_t off = 0;
    auto Aw = [&](size_t n) { float* p = base + off; off += n; return p; };
    float *q_u = Aw(NFu), *k_u = Aw(NFu), *v_u = Aw(NFu);
    float *q_i = Aw(NFi), *k_i = Aw(NFi), *v_i = Aw(NFi);
    float *tq_u = Aw(NFu), *sh_u = Aw(NFu), *tq_i = Aw(NFi), *sh_i = Aw(NFi);
    float *lraw_u = Aw(NFu), *lraw_i = Aw(NFi);
    float *meta_u = Aw(NFu), *meta_i = Aw(NFi);
    float *fused_u = Aw(NFu), *fused_i = Aw(NFi);
    float *ffn_u = Aw(NFu), *ffn_i = Aw(NFi);
    float *ffnh_u = Aw(NFu * 4), *ffnh_i = Aw(NFi * 4);
    float *sc = Aw((size_t)Emax * 8), *ex = Aw((size_t)Emax * 8);
    unsigned *mb = (unsigned*)Aw((size_t)NM * 8); float *zb = Aw((size_t)NM * 8);
    float *ms = Aw((size_t)Emax), *me = Aw((size_t)Emax);
    unsigned *mm = (unsigned*)Aw((size_t)NM); float *mz = Aw((size_t)NM);
    float *gate_u = Aw((size_t)NU * 4), *gate_i = Aw((size_t)NI * 4);
    float *gsum_u = Aw(F), *gsum_i = Aw(F), *gv_u = Aw(F), *gv_i = Aw(F);
    // phase-safe aliases (source buffers dead before these are written)
    float *local_u = q_u, *local_i = q_i, *mgate_u = tq_u, *mgate_i = tq_i;
    float *topo_u = k_u, *topo_i = k_i, *rg_u = sh_u, *rg_i = sh_i;

    const dim3 blk(128);
    auto gx = [](int nr) { return (unsigned)((nr + 15) / 16); };
    const float* Z = nullptr;

    // ---- Phase A: fused LN+GEMM projections (WMMA) ----
    lngemm_kernel<128,0,true><<<dim3(gx(NU),1),blk,0,stream>>>(h_u,F,Z,0,Z,0,Z,0, Pu.qg,Pu.qb, Pu.qW,F, Pu.qc,nullptr, q_u,F,NU);
    lngemm_kernel<128,0,true><<<dim3(gx(NU),1),blk,0,stream>>>(h_u,F,Z,0,Z,0,Z,0, Pu.kg,Pu.kb, Pu.kW,F, Pu.kc,relk+0, k_u,F,NU);
    lngemm_kernel<128,0,true><<<dim3(gx(NU),1),blk,0,stream>>>(h_u,F,Z,0,Z,0,Z,0, Pu.vg,Pu.vb, Pu.vW,F, Pu.vc,relv+0, v_u,F,NU);
    lngemm_kernel<128,0,true><<<dim3(gx(NU),1),blk,0,stream>>>(h_u,F,Z,0,Z,0,Z,0, Pu.mqg,Pu.mqb, Pu.mqW,F, Pu.mqc,nullptr, tq_u,F,NU);
    lngemm_kernel<128,0,true><<<dim3(gx(NU),1),blk,0,stream>>>(h_u,F,Z,0,Z,0,Z,0, Pu.mvg,Pu.mvb, Pu.mvW,F, Pu.mvc,nullptr, sh_u,F,NU);
    lngemm_kernel<128,0,true><<<dim3(gx(NI),1),blk,0,stream>>>(h_i,F,Z,0,Z,0,Z,0, Pi.qg,Pi.qb, Pi.qW,F, Pi.qc,nullptr, q_i,F,NI);
    lngemm_kernel<128,0,true><<<dim3(gx(NI),1),blk,0,stream>>>(h_i,F,Z,0,Z,0,Z,0, Pi.kg,Pi.kb, Pi.kW,F, Pi.kc,relk+F, k_i,F,NI);
    lngemm_kernel<128,0,true><<<dim3(gx(NI),1),blk,0,stream>>>(h_i,F,Z,0,Z,0,Z,0, Pi.vg,Pi.vb, Pi.vW,F, Pi.vc,relv+F, v_i,F,NI);
    lngemm_kernel<128,0,true><<<dim3(gx(NI),1),blk,0,stream>>>(h_i,F,Z,0,Z,0,Z,0, Pi.mqg,Pi.mqb, Pi.mqW,F, Pi.mqc,nullptr, tq_i,F,NI);
    lngemm_kernel<128,0,true><<<dim3(gx(NI),1),blk,0,stream>>>(h_i,F,Z,0,Z,0,Z,0, Pi.mvg,Pi.mvb, Pi.mvW,F, Pi.mvc,nullptr, sh_i,F,NI);

    // ---- zero accumulators ----
    hipMemsetAsync(lraw_u, 0, NFu * 4, stream); hipMemsetAsync(lraw_i, 0, NFi * 4, stream);
    hipMemsetAsync(meta_u, 0, NFu * 4, stream); hipMemsetAsync(meta_i, 0, NFi * 4, stream);
    hipMemsetAsync(gsum_u, 0, F * 4, stream);   hipMemsetAsync(gsum_i, 0, F * 4, stream);

    // ---- Phase B: relational attention ----
    // r1: item -> user
    hipMemsetAsync(mb, 0, (size_t)NM * 8 * 4, stream); hipMemsetAsync(zb, 0, (size_t)NM * 8 * 4, stream);
    edge_score_kernel<<<(E1*8+255)/256,256,0,stream>>>(k_i, q_u, r1s, r1d, relb+8, rels+1, sc, mb, E1);
    edge_exp_kernel  <<<(E1*8+255)/256,256,0,stream>>>(sc, r1d, mb, ex, zb, E1);
    edge_agg_kernel  <<<(unsigned)(((long long)E1*F+255)/256),256,0,stream>>>(v_i, ex, zb, r1s, r1d, lraw_u, E1);
    // r0: user -> item
    hipMemsetAsync(mb, 0, (size_t)NM * 8 * 4, stream); hipMemsetAsync(zb, 0, (size_t)NM * 8 * 4, stream);
    edge_score_kernel<<<(E0*8+255)/256,256,0,stream>>>(k_u, q_i, r0s, r0d, relb+0, rels+0, sc, mb, E0);
    edge_exp_kernel  <<<(E0*8+255)/256,256,0,stream>>>(sc, r0d, mb, ex, zb, E0);
    edge_agg_kernel  <<<(unsigned)(((long long)E0*F+255)/256),256,0,stream>>>(v_u, ex, zb, r0s, r0d, lraw_i, E0);

    // ---- Phase C: metapath attention ----
    hipMemsetAsync(mm, 0, (size_t)NM * 4, stream); hipMemsetAsync(mz, 0, (size_t)NM * 4, stream);
    meta_score_kernel<<<(EU+7)/8,256,0,stream>>>(sh_u, tq_u, mus, mud, ms, mm, EU);
    meta_exp_kernel  <<<(EU+255)/256,256,0,stream>>>(ms, mud, mm, me, mz, EU);
    meta_agg_kernel  <<<(unsigned)(((long long)EU*F+255)/256),256,0,stream>>>(sh_u, me, mz, mus, mud, meta_u, EU);
    hipMemsetAsync(mm, 0, (size_t)NM * 4, stream); hipMemsetAsync(mz, 0, (size_t)NM * 4, stream);
    meta_score_kernel<<<(EI+7)/8,256,0,stream>>>(sh_i, tq_i, mis, mid, ms, mm, EI);
    meta_exp_kernel  <<<(EI+255)/256,256,0,stream>>>(ms, mid, mm, me, mz, EI);
    meta_agg_kernel  <<<(unsigned)(((long long)EI*F+255)/256),256,0,stream>>>(sh_i, me, mz, mis, mid, meta_i, EI);

    // ---- Phase D: node update ----
    float* out = (float*)d_out;
    auto update = [&](const float* h, int N, const NP& P,
                      float* lraw, float* meta, float* local, float* mgate, float* topo,
                      float* rg, float* gate, float* gsum, float* gv,
                      float* fused, float* ffnh, float* ffn, float* outp) {
        colsum_kernel<<<256,128,0,stream>>>(h, gsum, N);
        glob_kernel<<<1,128,0,stream>>>(gsum, 1.f/(float)N, P.globg, P.globb, P.globW, P.globc, gv);
        lngemm_kernel<128,0,true ><<<dim3(gx(N),1),blk,0,stream>>>(lraw,F,Z,0,Z,0,Z,0, P.outg,P.outb, P.outW,F, P.outc,nullptr, local,F,N);
        lngemm_kernel<256,3,false><<<dim3(gx(N),1),blk,0,stream>>>(h,F,meta,F,Z,0,Z,0, nullptr,nullptr, P.mgW,F, P.mgc,nullptr, mgate,F,N);
        lngemm_kernel<128,0,true ><<<dim3(gx(N),1),blk,0,stream>>>(meta,F,Z,0,Z,0,Z,0, P.topog,P.topob, P.topoW,F, P.topoc,nullptr, topo,F,N);
        lngemm_kernel<512,1,false><<<dim3(gx(N),1),blk,0,stream>>>(local,F,mgate,F,topo,F,gv,0, nullptr,nullptr, P.bg1W,F, P.bg1c,nullptr, rg,F,N);
        gate4_kernel<<<(N+255)/256,256,0,stream>>>(rg, P.bg2W, P.bg2c, gate, N);
        mix_ln_kernel<<<N,128,0,stream>>>(h, local, mgate, topo, gv, gate, P.n1g, P.n1b, fused, N);
        lngemm_kernel<128,2,true ><<<dim3(gx(N),4),blk,0,stream>>>(fused,F,Z,0,Z,0,Z,0, P.f1g,P.f1b, P.f1W,512, P.f1c,nullptr, ffnh,512,N);
        lngemm_kernel<512,0,false><<<dim3(gx(N),1),blk,0,stream>>>(ffnh,512,ffnh+128,512,ffnh+256,512,ffnh+384,512, nullptr,nullptr, P.f2W,F, P.f2c,nullptr, ffn,F,N);
        add_ln_kernel<<<N,128,0,stream>>>(fused, ffn, P.n2g, P.n2b, outp, N);
    };
    update(h_u, NU, Pu, lraw_u, meta_u, local_u, mgate_u, topo_u, rg_u, gate_u, gsum_u, gv_u, fused_u, ffnh_u, ffn_u, out);
    update(h_i, NI, Pi, lraw_i, meta_i, local_i, mgate_i, topo_i, rg_i, gate_i, gsum_i, gv_i, fused_i, ffnh_i, ffn_i, out + NFu);
}
